// Deformable_attention_69973607187177
// MI455X (gfx1250) — matrix operand order repaired and checked by reading, not compiled, not run on hardware
//
#include <hip/hip_runtime.h>
#include <hip/hip_bf16.h>
#include <stdint.h>

#define NF  64
#define HH  192
#define WW  192
#define BB  2
#define HWP (HH*WW)
#define HP  194
#define WP  194

typedef __bf16 bf16_t;
typedef __attribute__((ext_vector_type(16))) __bf16 v16bf;
typedef __attribute__((ext_vector_type(8)))  __bf16 v8bf;
typedef __attribute__((ext_vector_type(8)))  float  v8f;
typedef __attribute__((ext_vector_type(4)))  float  v4f;

__device__ __forceinline__ bf16_t f2bf(float f) {
  uint32_t u = __float_as_uint(f);
  uint32_t r = (u + 0x7FFFu + ((u >> 16) & 1u)) >> 16;   // round-to-nearest-even
  return __builtin_bit_cast(bf16_t, (uint16_t)r);
}

// ---------------------------------------------------------------- utilities
__global__ void fill_zero_u32(uint32_t* __restrict__ p, long n) {
  long i = (long)blockIdx.x * blockDim.x + threadIdx.x;
  if (i < n) p[i] = 0u;
}

// Build padded channel-last bf16 input for conv1: X1[b][y+1][x+1][0..63]=query, [64..127]=key
__global__ void pack_qk_kernel(const float* __restrict__ q, const float* __restrict__ k,
                               bf16_t* __restrict__ X1) {
  int idx = blockIdx.x * blockDim.x + threadIdx.x;
  if (idx >= BB * HWP) return;
  int b = idx / HWP;
  int p = idx - b * HWP;
  int y = p / WW, x = p - y * WW;
  bf16_t* dst = X1 + ((size_t)(b * HP + y + 1) * WP + (x + 1)) * 128;
  const float* qb = q + (size_t)b * NF * HWP + p;
  const float* kb = k + (size_t)b * NF * HWP + p;
#pragma unroll 4
  for (int c = 0; c < NF; ++c) {
    dst[c]      = f2bf(qb[(size_t)c * HWP]);
    dst[64 + c] = f2bf(kb[(size_t)c * HWP]);
  }
}

// OIHW fp32 -> bf16 packed W[co][k], k = tap*Cin + ci (co zero-padded to CoutPad rows)
__global__ void pack_w_kernel(const float* __restrict__ w, bf16_t* __restrict__ dst,
                              int Cout, int Cin, int total) {
  int idx = blockIdx.x * blockDim.x + threadIdx.x;
  if (idx >= total) return;
  int K   = Cin * 9;
  int co  = idx / K;
  int k   = idx - co * K;
  int tap = k / Cin;
  int ci  = k - tap * Cin;
  float v = 0.f;
  if (co < Cout) v = w[(size_t)(co * Cin + ci) * 9 + tap];
  dst[idx] = f2bf(v);
}

// ------------------------------------------------ implicit-GEMM 3x3 conv via WMMA
// One wave computes a 16(Cout) x 16(pixel) tile, K = Cin*9 in steps of 32.
// MODE 0: out = bf16 padded channel-last [B][HP][WP][COSTRIDE], bias + leaky-relu.
// MODE 1: out = fp32 channel-last [B][H][W][COSTRIDE], bias only, co>=216 dropped.
template<int CIN, int COSTRIDE, int MODE>
__launch_bounds__(128)
__global__ void conv3x3_wmma_kernel(const bf16_t* __restrict__ X,
                                    const bf16_t* __restrict__ Wt,
                                    const float*  __restrict__ bias,
                                    void* __restrict__ outv) {
  constexpr int KTOT = CIN * 9;
  const int lane = threadIdx.x & 31;
  const int wv   = threadIdx.x >> 5;
  const int tile = blockIdx.x * 4 + wv;                 // 0..2303 (H * W/16)
  const int y    = tile / (WW / 16);
  const int x    = (tile % (WW / 16)) * 16 + (lane & 15);
  const int hi8  = (lane >> 4) * 8;                     // A K-offset / D row offset
  const int b16  = (lane >> 4) * 16;                    // unused; B K-offset is hi8
  const int cob  = blockIdx.y * 16;
  const int b    = blockIdx.z;

  const bf16_t* arow = Wt + (size_t)(cob + (lane & 15)) * KTOT + hi8;
  const bf16_t* xb   = X + (size_t)b * (HP * WP) * CIN;

  v8f acc = {};
#pragma unroll 4
  for (int kt = 0; kt < KTOT; kt += 32) {
    const int tap = kt / CIN;
    const int ci0 = kt - tap * CIN;
    const int dy  = tap / 3, dx = tap - dy * 3;
    // B fragment: column N = lane%16 -> pixel x; K split {0..7,16..23} / {8..15,24..31}
    const bf16_t* bp = xb + ((size_t)((y + dy) * WP + (x + dx))) * CIN + ci0 + hi8;
    v8bf b0 = *(const v8bf*)(bp);
    v8bf b1 = *(const v8bf*)(bp + 16);
    // A fragment: row M = cob + lane%16; K split {0..7,16..23} / {8..15,24..31}
    const bf16_t* ap = arow + kt;
    v8bf a0 = *(const v8bf*)(ap);
    v8bf a1 = *(const v8bf*)(ap + 16);
    v16bf Af = __builtin_shufflevector(a0, a1, 0,1,2,3,4,5,6,7,8,9,10,11,12,13,14,15);
    v16bf Bf = __builtin_shufflevector(b0, b1, 0,1,2,3,4,5,6,7,8,9,10,11,12,13,14,15);
    acc = __builtin_amdgcn_wmma_f32_16x16x32_bf16(false, Af, false, Bf, (short)0, acc,
                                                  false, false);
  }

  const int co0 = cob + hi8;  // this lane holds rows co0..co0+7 at column pixel x
  if (MODE == 0) {
    bf16_t* op = (bf16_t*)outv +
                 ((size_t)(b * HP + (y + 1)) * WP + (x + 1)) * COSTRIDE + co0;
    v8bf o;
#pragma unroll
    for (int i = 0; i < 8; ++i) {
      float v = acc[i] + bias[co0 + i];
      v = (v >= 0.f) ? v : 0.1f * v;
      o[i] = f2bf(v);
    }
    *(v8bf*)op = o;
  } else {
    if (co0 < 216) {
      float* op = (float*)outv + ((size_t)(b * HH + y) * WW + x) * COSTRIDE + co0;
      v4f r0, r1;
#pragma unroll
      for (int i = 0; i < 4; ++i) {
        r0[i] = acc[i]     + bias[co0 + i];
        r1[i] = acc[i + 4] + bias[co0 + 4 + i];
      }
      *(v4f*)op       = r0;
      *(v4f*)(op + 4) = r1;
    }
  }
}

// ------------------------------------------------ deformable gather + top-2 attention
__device__ __forceinline__ void bilin_tap(const float* __restrict__ base, float ys, float xs,
                                          float& r0, float& r1) {
  const float yf = floorf(ys), xf = floorf(xs);
  r0 = 0.f; r1 = 0.f;
#pragma unroll
  for (int dy = 0; dy < 2; ++dy) {
#pragma unroll
    for (int dx = 0; dx < 2; ++dx) {
      const float yi = yf + (float)dy;
      const float xi = xf + (float)dx;
      const float wgt = (1.f - fabsf(ys - yi)) * (1.f - fabsf(xs - xi));
      const bool valid = (yi >= 0.f) && (yi <= (float)(HH - 1)) &&
                         (xi >= 0.f) && (xi <= (float)(WW - 1));
      const float yc = fminf(fmaxf(yi, 0.f), (float)(HH - 1));
      const float xc = fminf(fmaxf(xi, 0.f), (float)(WW - 1));
      const int lin = (int)(yc * (float)WW + xc);
      const float wm = valid ? wgt : 0.f;
      r0 += base[lin] * wm;
      r1 += base[lin + HWP] * wm;
    }
  }
}

// One wave per pixel; lane owns channels {2*lane, 2*lane+1} (same deformable group).
__launch_bounds__(256)
__global__ void attn_topk_kernel(const float* __restrict__ om,
                                 const float* __restrict__ query,
                                 const float* __restrict__ key,
                                 const float* __restrict__ value,
                                 float* __restrict__ out) {
  const int lane = threadIdx.x & 31;
  const int pix  = blockIdx.x * 8 + (threadIdx.x >> 5);  // grid covers B*HW exactly
  const int b  = pix / HWP;
  const int p  = pix - b * HWP;
  const int yy = p / WW;
  const int xx = p - yy * WW;
  const int c0 = lane * 2;
  const int dg = c0 >> 3;                                // C/DG = 8 channels per group
  const float* omp = om + (size_t)pix * 216;
  const size_t cb  = (size_t)b * NF * HWP + (size_t)c0 * HWP;
  const float* kb  = key + cb;
  const float* vb  = value + cb;
  const float q0 = query[cb + p];
  const float q1 = query[cb + HWP + p];

  float rel[9], offy[9], offx[9], msk[9], kk0[9], kk1[9];
#pragma unroll
  for (int n = 0; n < 9; ++n) {
    const float oy = omp[dg * 9 + n];
    const float ox = omp[72 + dg * 9 + n];
    const float mm = 1.f / (1.f + __expf(-omp[144 + dg * 9 + n]));
    offy[n] = oy; offx[n] = ox; msk[n] = mm;
    const float ys = (float)(yy + n / 3 - 1) + oy;
    const float xs = (float)(xx + n % 3 - 1) + ox;
    float a0, a1;
    bilin_tap(kb, ys, xs, a0, a1);
    a0 *= mm; a1 *= mm;
    kk0[n] = a0; kk1[n] = a1;
    float r = a0 * q0 + a1 * q1;                         // partial over 2 channels
    r += __shfl_xor(r, 16, 32); r += __shfl_xor(r, 8, 32);
    r += __shfl_xor(r, 4, 32);  r += __shfl_xor(r, 2, 32);
    r += __shfl_xor(r, 1, 32);
    rel[n] = r;                                          // full sum over C=64
  }

  // top-2 (first-occurrence ties, matching lax.top_k)
  int i1 = 0; float v1 = rel[0];
#pragma unroll
  for (int n = 1; n < 9; ++n) { if (rel[n] > v1) { v1 = rel[n]; i1 = n; } }
  int i2 = (i1 == 0) ? 1 : 0; float v2 = rel[i2];
#pragma unroll
  for (int n = 0; n < 9; ++n) { if (n != i1 && rel[n] > v2) { v2 = rel[n]; i2 = n; } }
  const float e  = __expf(v2 - v1);
  const float s  = 1.f / (1.f + e);
  const float cA = s, cB = e * s;                        // softmax([v1, v2])

  // weight = q . (cA*k[i1] + cB*k[i2])
  const float ku0 = cA * kk0[i1] + cB * kk0[i2];
  const float ku1 = cA * kk1[i1] + cB * kk1[i2];
  float wr = q0 * ku0 + q1 * ku1;
  wr += __shfl_xor(wr, 16, 32); wr += __shfl_xor(wr, 8, 32);
  wr += __shfl_xor(wr, 4, 32);  wr += __shfl_xor(wr, 2, 32);
  wr += __shfl_xor(wr, 1, 32);
  if (lane == 0) out[pix] = wr;

  // value gathers only for the two selected taps
  float a0, a1, b0, b1;
  float ys = (float)(yy + i1 / 3 - 1) + offy[i1];
  float xs = (float)(xx + i1 % 3 - 1) + offx[i1];
  bilin_tap(vb, ys, xs, a0, a1);
  ys = (float)(yy + i2 / 3 - 1) + offy[i2];
  xs = (float)(xx + i2 % 3 - 1) + offx[i2];
  bilin_tap(vb, ys, xs, b0, b1);
  const float vo0 = cA * msk[i1] * a0 + cB * msk[i2] * b0;
  const float vo1 = cA * msk[i1] * a1 + cB * msk[i2] * b1;

  float* vout = out + (size_t)BB * HWP + cb + p;         // v_out after weight block
  vout[0]   = vo0;
  vout[HWP] = vo1;
}

// ---------------------------------------------------------------- launcher
extern "C" void kernel_launch(void* const* d_in, const int* in_sizes, int n_in,
                              void* d_out, int out_size, void* d_ws, size_t ws_size,
                              hipStream_t stream) {
  (void)in_sizes; (void)n_in; (void)out_size; (void)ws_size;
  const float* query = (const float*)d_in[0];
  const float* keyt  = (const float*)d_in[1];
  const float* value = (const float*)d_in[2];
  const float* w1    = (const float*)d_in[3];
  const float* b1    = (const float*)d_in[4];
  const float* w2    = (const float*)d_in[5];
  const float* b2    = (const float*)d_in[6];
  const float* w_om  = (const float*)d_in[7];
  const float* b_om  = (const float*)d_in[8];

  char* ws = (char*)d_ws;
  size_t off = 0;
  auto carve = [&](size_t bytes) -> void* {
    void* p = (void*)(ws + off);
    off += (bytes + 255) & ~(size_t)255;
    return p;
  };
  const size_t nX1 = (size_t)BB * HP * WP * 128;   // concat(q,k) padded, bf16
  const size_t nX2 = (size_t)BB * HP * WP * 64;    // feat buffers padded, bf16
  const size_t nOM = (size_t)BB * HWP * 216;       // offsets/masks, fp32 channel-last
  bf16_t* X1   = (bf16_t*)carve(nX1 * 2);
  bf16_t* X2   = (bf16_t*)carve(nX2 * 2);
  bf16_t* X3   = (bf16_t*)carve(nX2 * 2);
  float*  OM   = (float*) carve(nOM * 4);
  bf16_t* W1p  = (bf16_t*)carve((size_t)64  * 1152 * 2);
  bf16_t* W2p  = (bf16_t*)carve((size_t)64  * 576  * 2);
  bf16_t* WOMp = (bf16_t*)carve((size_t)224 * 576  * 2);

  // zero padded feature buffers (borders must read as 0 for SAME conv)
  { long n = (long)(nX1 / 2); fill_zero_u32<<<(int)((n + 255) / 256), 256, 0, stream>>>((uint32_t*)X1, n); }
  { long n = (long)(nX2 / 2); fill_zero_u32<<<(int)((n + 255) / 256), 256, 0, stream>>>((uint32_t*)X2, n); }
  { long n = (long)(nX2 / 2); fill_zero_u32<<<(int)((n + 255) / 256), 256, 0, stream>>>((uint32_t*)X3, n); }

  pack_qk_kernel<<<(BB * HWP + 255) / 256, 256, 0, stream>>>(query, keyt, X1);
  { int n = 64 * 1152; pack_w_kernel<<<(n + 255) / 256, 256, 0, stream>>>(w1,   W1p, 64, 128, n); }
  { int n = 64 * 576;  pack_w_kernel<<<(n + 255) / 256, 256, 0, stream>>>(w2,   W2p, 64, 64,  n); }
  { int n = 224 * 576; pack_w_kernel<<<(n + 255) / 256, 256, 0, stream>>>(w_om, WOMp, 216, 64, n); }

  // H * W/16 = 2304 pixel tiles per batch, 4 waves per 128-thread block -> grid.x = 576
  dim3 blk(128);
  conv3x3_wmma_kernel<128, 64, 0><<<dim3(576, 4, BB),  blk, 0, stream>>>(X1, W1p,  b1,   (void*)X2);
  conv3x3_wmma_kernel< 64, 64, 0><<<dim3(576, 4, BB),  blk, 0, stream>>>(X2, W2p,  b2,   (void*)X3);
  conv3x3_wmma_kernel< 64, 216,1><<<dim3(576, 14, BB), blk, 0, stream>>>(X3, WOMp, b_om, (void*)OM);

  attn_topk_kernel<<<BB * HWP / 8, 256, 0, stream>>>(OM, query, keyt, value, (float*)d_out);
}
